// MultiHeadAttention_51668456571097
// MI455X (gfx1250) — compile-verified, matrix operand order checked by reading, not run on hardware
//
#include <hip/hip_runtime.h>
#include <hip/hip_bf16.h>

// MI455X / gfx1250: wave32, v_wmma_f32_16x16x32_f16, TDM tensor_load_to_lds staging.

typedef __attribute__((ext_vector_type(16))) _Float16 v16h;
typedef __attribute__((ext_vector_type(8)))  float    v8f;
typedef __attribute__((ext_vector_type(4)))  unsigned int v4u;
typedef __attribute__((ext_vector_type(8)))  int      v8i;
typedef __attribute__((ext_vector_type(4)))  int      v4i;

constexpr int Bc  = 16;   // batch
constexpr int Lc  = 512;  // sequence length
constexpr int LAc = 8;    // aspect length
constexpr int Dc  = 768;  // model dim
constexpr int Hc  = 12;   // heads
constexpr int DKc = 64;   // head dim

#if defined(__has_builtin)
#if __has_builtin(__builtin_amdgcn_tensor_load_to_lds) && __has_builtin(__builtin_amdgcn_s_wait_tensorcnt)
#define HAVE_TDM 1
#endif
#endif

// K index inside a 16x32 f16 A/B fragment for element e (0..15) of lane-group g (0..1).
// Per CDNA5 ISA 7.12.2: VGPR0..3 hold K={0..7}(+8g), VGPR4..7 hold K={16..23}(+8g).
__device__ __forceinline__ int kmap(int e, int g) {
  return (e < 8 ? e : e + 8) + 8 * g;
}

__device__ __forceinline__ v8f wmma16(v16h a, v16h b, v8f c) {
  return __builtin_amdgcn_wmma_f32_16x16x32_f16(false, a, false, b, (short)0, c, false, false);
}

// ---- TDM: DMA a rows x cols f16 panel (contiguous, stride==cols) into LDS ---
// Descriptor per CDNA5 ISA ch.8 (D# group0/group1). Issued by one wave;
// completion via s_wait_tensorcnt, then workgroup barrier.
__device__ __forceinline__ void tdm_load_panel_f16(const _Float16* gsrc, void* lds_dst,
                                                   int rows, int cols) {
#if defined(HAVE_TDM)
  const unsigned lds_off = (unsigned)(size_t)lds_dst;        // generic LDS addr[31:0] == offset
  const unsigned long long ga = (unsigned long long)(size_t)gsrc;

  v4u g0;
  g0[0] = 1u;                                                // count=1 (valid descriptor)
  g0[1] = lds_off;                                           // lds_addr  [63:32]
  g0[2] = (unsigned)(ga & 0xffffffffull);                    // global_addr lo [95:64]
  g0[3] = (unsigned)((ga >> 32) & 0x1ffffffull)              // global_addr hi [120:96]
        | (2u << 30);                                        // type=2 ("image") [127:126]

  v8i g1;
  g1[0] = (1 << 16);                                         // data_size=1 (2B); mask/pad=0
  g1[1] = (cols & 0xffff) << 16;                             // tensor_dim0 lo16 @ [63:48]
  g1[2] = ((cols >> 16) & 0xffff) | ((rows & 0xffff) << 16); // dim0 hi | tensor_dim1 lo
  g1[3] = ((rows >> 16) & 0xffff) | ((cols & 0xffff) << 16); // dim1 hi | tile_dim0
  g1[4] = (rows & 0xffff);                                   // tile_dim1 (tile_dim2=0)
  g1[5] = cols;                                              // tensor_dim0_stride lo32
  g1[6] = 0;                                                 // stride hi | dim1_stride lo
  g1[7] = 0;

  v4i z4 = {0, 0, 0, 0};
#if __clang_major__ >= 23
  v8i z8 = {0, 0, 0, 0, 0, 0, 0, 0};
  __builtin_amdgcn_tensor_load_to_lds(g0, g1, z4, z4, z8, 0);
#else
  __builtin_amdgcn_tensor_load_to_lds(g0, g1, z4, z4, 0);
#endif
  __builtin_amdgcn_s_wait_tensorcnt(0);
#else
  (void)gsrc; (void)lds_dst; (void)rows; (void)cols;
#endif
}

// ---- fragment loaders -------------------------------------------------------

__device__ __forceinline__ v16h load_a_f32(const float* __restrict__ A,
                                           int row0, int k0, int lda) {
  int lane = threadIdx.x & 31;
  int m = lane & 15, g = lane >> 4;
  const float* base = A + (size_t)(row0 + m) * lda + k0;
  v16h a;
#pragma unroll
  for (int e = 0; e < 16; ++e) a[e] = (_Float16)base[kmap(e, g)];
  return a;
}

__device__ __forceinline__ v16h load_b_f32(const float* __restrict__ Bm,
                                           int k0, int col0, int ldb) {
  int lane = threadIdx.x & 31;
  int n = lane & 15, g = lane >> 4;
  v16h b;
#pragma unroll
  for (int e = 0; e < 16; ++e)
    b[e] = (_Float16)Bm[(size_t)(k0 + kmap(e, g)) * ldb + col0 + n];
  return b;
}

__device__ __forceinline__ v16h load_a_f16(const _Float16* __restrict__ A,
                                           int row0, int k0, int lda) {
  int lane = threadIdx.x & 31;
  int m = lane & 15, g = lane >> 4;
  const _Float16* base = A + (size_t)(row0 + m) * lda + k0;
  v16h a;
#pragma unroll
  for (int e = 0; e < 16; ++e) a[e] = base[kmap(e, g)];
  return a;
}

__device__ __forceinline__ v16h load_b_f16(const _Float16* __restrict__ Bm,
                                           int k0, int col0, int ldb) {
  int lane = threadIdx.x & 31;
  int n = lane & 15, g = lane >> 4;
  v16h b;
#pragma unroll
  for (int e = 0; e < 16; ++e)
    b[e] = Bm[(size_t)(k0 + kmap(e, g)) * ldb + col0 + n];
  return b;
}

// ---- kernel 1: fused projection  out = f16(X @ W + bias) --------------------
// layout 0: out[((b*H+h)*rows_pb + r)*DK + dk]   (Q / aspect head layout)
// layout 1: out[((b*H+h)*DK + dk)*rows_pb + r]   (K transposed: scores B-frags row-major)
__global__ void proj_wmma(const float* __restrict__ X, const float* __restrict__ W,
                          const float* __restrict__ bias, _Float16* __restrict__ out,
                          int rows_pb, int layout) {
  const int row0 = blockIdx.x * 16;
  const int col0 = blockIdx.y * 16;

  v8f c = {};
#pragma unroll 4
  for (int k0 = 0; k0 < Dc; k0 += 32) {
    if (k0 + 32 < Dc)  // prefetch next weight panel -> global_prefetch_b8
      __builtin_prefetch(&W[(size_t)(k0 + 32) * Dc + col0], 0, 0);
    v16h a = load_a_f32(X, row0, k0, Dc);
    v16h b = load_b_f32(W, k0, col0, Dc);
    c = wmma16(a, b, c);
  }

  const int lane = threadIdx.x & 31;
  const int n = lane & 15, g = lane >> 4;
  const int col = col0 + n;
  const float bv = bias ? bias[col] : 0.0f;
  const int h = col >> 6;
  const int dk = col & 63;

#pragma unroll
  for (int r = 0; r < 8; ++r) {
    const int R = row0 + r + 8 * g;
    const int bidx = R / rows_pb;
    const int ridx = R % rows_pb;
    const _Float16 v = (_Float16)(c[r] + bv);
    size_t o;
    if (layout == 0)
      o = ((size_t)(bidx * Hc + h) * rows_pb + ridx) * DKc + dk;
    else
      o = ((size_t)(bidx * Hc + h) * DKc + dk) * rows_pb + ridx;
    out[o] = v;
  }
}

// ---- kernel 2: scores + masked softmax --------------------------------------
// One 8-wave workgroup per (b*H+h, 16-row tile). K panel staged into LDS via TDM
// in two 32x512 f16 stages; panel buffer is reused as the f32 softmax strip.
__global__ void __launch_bounds__(256)
scores_softmax(const _Float16* __restrict__ Qh, const _Float16* __restrict__ Kt,
               const int* __restrict__ mask, float* __restrict__ out) {
  __shared__ float smbuf[16 * (Lc + 4)];     // 33 KB: aliased as f16 panel then f32 strip
  __shared__ float red[16][16];

  const int bh = blockIdx.x;        // 0..B*H-1
  const int i0 = blockIdx.y * 16;   // row tile
  const int wave = threadIdx.x >> 5;
  const int lane = threadIdx.x & 31;
  const int n = lane & 15, g = lane >> 4;

  const _Float16* Qbase = Qh + (size_t)bh * Lc * DKc;   // [L, DK]
  const _Float16* Kbase = Kt + (size_t)bh * DKc * Lc;   // [DK, L]
  _Float16* panel = reinterpret_cast<_Float16*>(smbuf); // [32][Lc] per stage

  // Q A-fragments: one per K-stage, reused by all 4 column tiles of this wave.
  v16h afrag0 = load_a_f16(Qbase, i0, 0, DKc);
  v16h afrag1 = load_a_f16(Qbase, i0, 32, DKc);
  v8f acc[4] = {};

#pragma unroll
  for (int ks = 0; ks < 2; ++ks) {
    __syncthreads();                      // previous panel fully consumed
#if defined(HAVE_TDM)
    if (wave == 0)
      tdm_load_panel_f16(Kbase + (size_t)(ks * 32) * Lc, panel, 32, Lc);
#else
    {                                     // cooperative 16B copy fallback
      const uint4* src = (const uint4*)(Kbase + (size_t)(ks * 32) * Lc);
      uint4* dst = (uint4*)panel;
      for (int t = threadIdx.x; t < (32 * Lc) / 8; t += 256) dst[t] = src[t];
    }
#endif
    __syncthreads();                      // panel visible to all waves
#pragma unroll
    for (int t = 0; t < 4; ++t) {
      const int j0 = (wave * 4 + t) * 16;
      v16h b;
#pragma unroll
      for (int e = 0; e < 16; ++e) b[e] = panel[kmap(e, g) * Lc + j0 + n];
      acc[t] = wmma16(ks == 0 ? afrag0 : afrag1, b, acc[t]);
    }
  }
  __syncthreads();                        // done with panel; reuse as strip

  float (*sm)[Lc + 4] = reinterpret_cast<float (*)[Lc + 4]>(smbuf);
#pragma unroll
  for (int t = 0; t < 4; ++t) {
    const int j0 = (wave * 4 + t) * 16;
#pragma unroll
    for (int r = 0; r < 8; ++r)
      sm[r + 8 * g][j0 + n] = acc[t][r] * 0.125f;   // 1/sqrt(DK)
  }
  __syncthreads();

  // masked row softmax: row = tid>>4, each of 16 segment-threads owns 32 cols
  const int row = threadIdx.x >> 4;
  const int seg = threadIdx.x & 15;
  const int b = bh / Hc;
  const int i = i0 + row;
  const int* mrow = mask + ((size_t)b * Lc + i) * Lc;

  float mx = -1e30f;
#pragma unroll 4
  for (int j = seg * 32; j < seg * 32 + 32; ++j) {
    float v = (mrow[j] == 0) ? -1e9f : sm[row][j];
    sm[row][j] = v;
    mx = fmaxf(mx, v);
  }
  red[row][seg] = mx;
  __syncthreads();
  if (seg == 0) {
    float m2 = red[row][0];
#pragma unroll
    for (int t = 1; t < 16; ++t) m2 = fmaxf(m2, red[row][t]);
    red[row][0] = m2;
  }
  __syncthreads();
  const float rmax = red[row][0];

  float s = 0.0f;
#pragma unroll 4
  for (int j = seg * 32; j < seg * 32 + 32; ++j) {
    float e = __expf(sm[row][j] - rmax);
    sm[row][j] = e;
    s += e;
  }
  __syncthreads();
  red[row][seg] = s;
  __syncthreads();
  if (seg == 0) {
    float t2 = 0.0f;
#pragma unroll
    for (int t = 0; t < 16; ++t) t2 += red[row][t];
    red[row][0] = t2;
  }
  __syncthreads();
  const float inv = 1.0f / red[row][0];

  float* orow = out + ((size_t)bh * Lc + i) * Lc + seg * 32;
#pragma unroll
  for (int u = 0; u < 32; u += 4) {
    float4 v4;
    v4.x = sm[row][seg * 32 + u + 0] * inv;
    v4.y = sm[row][seg * 32 + u + 1] * inv;
    v4.z = sm[row][seg * 32 + u + 2] * inv;
    v4.w = sm[row][seg * 32 + u + 3] * inv;
    *(float4*)(orow + u) = v4;
  }
}

// ---- kernel 3: aspect scores  sigmoid((a . k + bias) * am) ------------------
__global__ void aspect_kernel(const _Float16* __restrict__ Ah, const _Float16* __restrict__ Kt,
                              const float* __restrict__ amask, const float* __restrict__ bias,
                              float* __restrict__ out) {
  const int bh = blockIdx.x;
  const int j0 = blockIdx.y * 16;
  const int lane = threadIdx.x & 31;
  const int m = lane & 15, g = lane >> 4, n = m;

  const _Float16* Abase = Ah + (size_t)bh * LAc * DKc;  // [LA, DK]
  const _Float16* Kbase = Kt + (size_t)bh * DKc * Lc;   // [DK, L]

  v16h a0, a1;
#pragma unroll
  for (int e = 0; e < 16; ++e) {
    if (m < LAc) {
      a0[e] = Abase[(size_t)m * DKc + kmap(e, g)];
      a1[e] = Abase[(size_t)m * DKc + 32 + kmap(e, g)];
    } else {
      a0[e] = (_Float16)0.0f;
      a1[e] = (_Float16)0.0f;
    }
  }

  v8f c = {};
  c = wmma16(a0, load_b_f16(Kbase, 0, j0, Lc), c);
  c = wmma16(a1, load_b_f16(Kbase, 32, j0, Lc), c);

  const int b = bh / Hc;
  const float bv = bias[0];
  const int col = j0 + n;
  if (g == 0) {                       // C-frag lanes 0..15 hold rows 0..7 (valid LA rows)
#pragma unroll
    for (int r = 0; r < 8; ++r) {
      const float am = amask[b * LAc + r];
      const float x = (c[r] + bv) * am;
      out[((size_t)bh * LAc + r) * Lc + col] = 1.0f / (1.0f + __expf(-x));
    }
  }
}

// ---- launch -----------------------------------------------------------------
extern "C" void kernel_launch(void* const* d_in, const int* in_sizes, int n_in,
                              void* d_out, int out_size, void* d_ws, size_t ws_size,
                              hipStream_t stream) {
  (void)in_sizes; (void)n_in; (void)out_size; (void)ws_size;

  const float* query       = (const float*)d_in[0];
  const float* key         = (const float*)d_in[1];
  const int*   mask        = (const int*)d_in[2];
  const float* aspect      = (const float*)d_in[3];
  const float* aspect_mask = (const float*)d_in[4];
  const float* Wq          = (const float*)d_in[5];
  const float* bq          = (const float*)d_in[6];
  const float* Wk          = (const float*)d_in[7];
  const float* bk          = (const float*)d_in[8];
  const float* Wa          = (const float*)d_in[9];
  const float* bias        = (const float*)d_in[10];

  _Float16* Qh = (_Float16*)d_ws;                       // [B,H,L,DK] f16
  _Float16* Kt = Qh + (size_t)Bc * Hc * Lc * DKc;       // [B,H,DK,L] f16
  _Float16* Ah = Kt + (size_t)Bc * Hc * DKc * Lc;       // [B,H,LA,DK] f16

  float* out_aspect = (float*)d_out;                                  // [B,H,LA,L]
  float* out_attn   = out_aspect + (size_t)Bc * Hc * LAc * Lc;        // [B,H,L,L]

  proj_wmma<<<dim3((Bc * Lc) / 16, Dc / 16), 32, 0, stream>>>(query, Wq, bq, Qh, Lc, 0);
  proj_wmma<<<dim3((Bc * Lc) / 16, Dc / 16), 32, 0, stream>>>(key, Wk, bk, Kt, Lc, 1);
  proj_wmma<<<dim3((Bc * LAc) / 16, Dc / 16), 32, 0, stream>>>(aspect, Wa, nullptr, Ah, LAc, 0);

  scores_softmax<<<dim3(Bc * Hc, Lc / 16), 256, 0, stream>>>(Qh, Kt, mask, out_attn);
  aspect_kernel<<<dim3(Bc * Hc, Lc / 16), 32, 0, stream>>>(Ah, Kt, aspect_mask, bias, out_aspect);
}
